// RNNPredictor_64493228917206
// MI455X (gfx1250) — matrix operand order repaired and checked by reading, not compiled, not run on hardware
//
#include <hip/hip_runtime.h>

#define B_ 256
#define T_ 256
#define I_ 128
#define H_ 512
#define HPAD 520   // LDS h row padding: stride 1040B -> rows shift 4 banks, conflict-free b128 reads
#define XPAD 516   // LDS xs row padding: stride 2064B -> half-wave groups hit disjoint banks

#if __has_builtin(__builtin_amdgcn_global_load_async_to_lds_b128) && \
    __has_builtin(__builtin_amdgcn_s_wait_asynccnt)
#define USE_ASYNC 1
#else
#define USE_ASYNC 0
#endif

typedef __attribute__((ext_vector_type(16))) __bf16 v16bf;
typedef __attribute__((ext_vector_type(8)))  float  v8f;
typedef __attribute__((ext_vector_type(4)))  int    v4i;

union V16 { v16bf v; uint4 q[2]; };

__device__ __forceinline__ float fast_tanh(float x) {
#if __has_builtin(__builtin_amdgcn_tanhf)
  return __builtin_amdgcn_tanhf(x);   // v_tanh_f32 (trans op)
#else
  return tanhf(x);
#endif
}

// 16B async copy global -> LDS (ASYNCcnt), with synchronous fallback.
__device__ __forceinline__ void stage16(const float* gsrc, float* ldst) {
#if USE_ASYNC
  v4i* gs = (v4i*)gsrc;   // strip const, retype (generic AS)
  v4i* ls = (v4i*)ldst;
  __builtin_amdgcn_global_load_async_to_lds_b128(
      (__attribute__((address_space(1))) v4i*)gs,
      (__attribute__((address_space(3))) v4i*)ls, 0, 0);
#else
  *(float4*)ldst = *(const float4*)gsrc;
#endif
}
__device__ __forceinline__ void async_drain() {
#if USE_ASYNC
  __builtin_amdgcn_s_wait_asynccnt(0);
#endif
}

// ---------------------------------------------------------------------------
// Pack a row-major [N, K=KC*32] f32 weight into the WMMA bf16 B-operand layout:
// flat = ((nt*KC + kc)*32 + lane)*16 + j ; lane holds K runs [half*8, +8) and
// [16+half*8, +8) (offset kc*32) for column n = nt*16 + (lane&15), half = lane>>4.
// ---------------------------------------------------------------------------
template <int KC>
__global__ void pack_b_kernel(const float* __restrict__ src,
                              __bf16* __restrict__ dst, int total) {
  int tid = blockIdx.x * blockDim.x + threadIdx.x;
  if (tid >= total) return;
  int j    = tid & 15;
  int lane = (tid >> 4) & 31;
  int rest = tid >> 9;
  int kc   = rest & (KC - 1);          // KC is a power of two
  int nt   = rest / KC;
  int n = nt * 16 + (lane & 15);
  int k = kc * 32 + ((j < 8) ? ((lane >> 4) * 8 + j) : (16 + (lane >> 4) * 8 + (j - 8)));
  dst[tid] = (__bf16)src[(size_t)n * (KC * 32) + k];
}

// ---------------------------------------------------------------------------
// Pack x (f32, rows = b*T+t of length I=KC*32) into the WMMA bf16 A layout:
// flat = ((mt*KC + kc)*32 + lane)*16 + j ; per-lane K runs [half*8, +8) and
// [16+half*8, +8) per ISA 7.12.2 16-bit A map.
// ---------------------------------------------------------------------------
template <int KC>
__global__ void pack_a_kernel(const float* __restrict__ src,
                              __bf16* __restrict__ dst, int total) {
  int tid = blockIdx.x * blockDim.x + threadIdx.x;
  if (tid >= total) return;
  int j    = tid & 15;
  int lane = (tid >> 4) & 31;
  int rest = tid >> 9;
  int kc   = rest & (KC - 1);
  int mt   = rest / KC;
  int m    = lane & 15;
  int half = lane >> 4;
  int k = kc * 32 + ((j < 8) ? (half * 8 + j) : (16 + half * 8 + (j - 8)));
  dst[tid] = (__bf16)src[(size_t)(mt * 16 + m) * (KC * 32) + k];
}

__global__ void bias_kernel(const float* __restrict__ b_ih,
                            const float* __restrict__ b_hh,
                            float* __restrict__ bsum) {
  int i = blockIdx.x * blockDim.x + threadIdx.x;
  if (i < H_) bsum[i] = b_ih[i] + b_hh[i];
}

// ---------------------------------------------------------------------------
// xs[t][b][n] = sum_k x[b][t][k] * W_ih[n][k] + (b_ih+b_hh)[n]
// One wave per 16x16 tile; operands pre-packed bf16 -> inner loop is pure
// global_load_b128 + v_wmma; epilogue uses immediate-offset stores.
// ---------------------------------------------------------------------------
__global__ void __launch_bounds__(256)
xs_gemm_kernel(const __bf16* __restrict__ xp,   // packed x (A layout)
               const __bf16* __restrict__ wp,   // packed W_ih (B layout)
               const float* __restrict__ bsum,
               float* __restrict__ xs) {
  const int lane = threadIdx.x & 31;
  const int wid  = blockIdx.x * (blockDim.x >> 5) + (threadIdx.x >> 5);
  const int nt   = wid & 31;     // H/16 = 32 tiles
  const int mt   = wid >> 5;     // (B*T)/16 = 4096 tiles
  const int half = lane >> 4;
  const int l16  = lane & 15;

  const __bf16* ab = xp + (size_t)mt * (4 * 32 * 16) + lane * 16;
  const __bf16* bb = wp + (size_t)nt * (4 * 32 * 16) + lane * 16;

  v8f acc = {};
  #pragma unroll
  for (int kc = 0; kc < 4; ++kc) {
    V16 av, bv;
    av.q[0] = *(const uint4*)(ab + kc * 512);
    av.q[1] = *(const uint4*)(ab + kc * 512 + 8);
    bv.q[0] = *(const uint4*)(bb + kc * 512);
    bv.q[1] = *(const uint4*)(bb + kc * 512 + 8);
    acc = __builtin_amdgcn_wmma_f32_16x16x32_bf16(false, av.v, false, bv.v,
                                                  (short)0, acc, false, false);
  }
  const int ncol = nt * 16 + l16;
  const float bias = bsum[ncol];
  // row = mt*16 + r + 8*half = b*T + t ; tiles are T-aligned so b is fixed
  const int b  = (mt * 16) >> 8;
  const int t0 = ((mt * 16) & 255) + 8 * half;
  float* xsb = xs + (size_t)(t0 * B_ + b) * H_ + ncol;
  #pragma unroll
  for (int r = 0; r < 8; ++r)
    xsb[r * (B_ * H_)] = acc[r] + bias;   // immediate offsets r*512KB
}

// ---------------------------------------------------------------------------
// Persistent recurrence: block = one 16-row batch tile, 16 waves x 2 N-tiles.
// h double-buffered in LDS (bf16); W_hh pre-packed bf16 streamed from L2;
// xs tiles double-buffered in LDS via GLOBAL_LOAD_ASYNC_TO_LDS_B128, the DMA
// for step t+1 overlapping step t's 32-WMMA chain (drained with
// s_wait_asynccnt 0 before the end-of-step barrier).
// ---------------------------------------------------------------------------
__global__ void __launch_bounds__(512, 1)
rnn_kernel(const float* __restrict__ xs,
           const __bf16* __restrict__ wp,   // packed W_hh (B layout)
           const float* __restrict__ fcw,
           const float* __restrict__ fcb,
           float* __restrict__ out) {
  __shared__ __align__(16) __bf16 hbuf[2][16][HPAD];
  __shared__ __align__(16) float  xbuf[2][16][XPAD];

  const int tid   = threadIdx.x;
  const int lane  = tid & 31;
  const int wave  = tid >> 5;
  const int half  = lane >> 4;
  const int l16   = lane & 15;
  const int bbase = blockIdx.x * 16;
  const int nt0   = wave * 2;

  for (int i = tid; i < 16 * HPAD; i += 512) hbuf[0][0][i] = (__bf16)0.0f;  // h0 = 0

  // Loop-invariant bases (immediate offsets cover kc / r / tile1):
  const __bf16* wb0 = wp + (size_t)nt0       * (16 * 32 * 16) + lane * 16;
  const __bf16* wb1 = wp + (size_t)(nt0 + 1) * (16 * 32 * 16) + lane * 16;
  const __bf16* ar[2] = { &hbuf[0][l16][half * 8], &hbuf[1][l16][half * 8] };
  __bf16*       hw[2] = { &hbuf[0][8 * half][nt0 * 16 + l16],
                          &hbuf[1][8 * half][nt0 * 16 + l16] };
  const float*  xr[2] = { &xbuf[0][8 * half][nt0 * 16 + l16],
                          &xbuf[1][8 * half][nt0 * 16 + l16] };

  // Per-thread staging slice: 4 rows x 16B, fully coalesced across the block.
  const int sc = (tid & 127) * 4;   // column (floats)
  const int sr = tid >> 7;          // base row 0..3
  const float* xg = xs + (size_t)bbase * H_;   // step-t tile base (t = 0)

  // Prologue: stage t=0, drain, publish (also covers the h0 zero-fill).
  #pragma unroll
  for (int i = 0; i < 4; ++i)
    stage16(xg + (size_t)(sr + 4 * i) * H_ + sc, &xbuf[0][sr + 4 * i][sc]);
  async_drain();
  __syncthreads();

  int cur = 0;
  for (int t = 0; t < T_; ++t) {
    if (t + 1 < T_) {  // kick off next step's xs DMA; overlaps the k-loop below
      const float* xn = xg + (size_t)(B_ * H_);
      #pragma unroll
      for (int i = 0; i < 4; ++i)
        stage16(xn + (size_t)(sr + 4 * i) * H_ + sc, &xbuf[(t + 1) & 1][sr + 4 * i][sc]);
    }

    const __bf16* a = ar[cur];
    v8f acc0 = {}, acc1 = {};
    #pragma unroll 4
    for (int kc = 0; kc < 16; ++kc) {
      V16 av, b0, b1;
      av.q[0] = *(const uint4*)(a + kc * 32);        // ISA 16-bit A layout
      av.q[1] = *(const uint4*)(a + kc * 32 + 16);
      b0.q[0] = *(const uint4*)(wb0 + kc * 512);
      b0.q[1] = *(const uint4*)(wb0 + kc * 512 + 8);
      b1.q[0] = *(const uint4*)(wb1 + kc * 512);
      b1.q[1] = *(const uint4*)(wb1 + kc * 512 + 8);
      acc0 = __builtin_amdgcn_wmma_f32_16x16x32_bf16(false, av.v, false, b0.v,
                                                     (short)0, acc0, false, false);
      acc1 = __builtin_amdgcn_wmma_f32_16x16x32_bf16(false, av.v, false, b1.v,
                                                     (short)0, acc1, false, false);
    }

    const float* xc = xr[t & 1];      // this step's staged xs tile (LDS)
    __bf16* hn = hw[cur ^ 1];
    #pragma unroll
    for (int r = 0; r < 8; ++r) {
      float v0 = acc0[r] + xc[r * XPAD];        // immediate r*2064B ds offsets
      float v1 = acc1[r] + xc[r * XPAD + 16];   // tile1: +64B
      hn[r * HPAD]      = (__bf16)fast_tanh(v0);
      hn[r * HPAD + 16] = (__bf16)fast_tanh(v1);
    }
    async_drain();     // next step's tile fully landed (overlapped with k-loop)
    __syncthreads();   // publish h(t+1) and xs(t+1) across all 16 waves
    cur ^= 1;
    xg += B_ * H_;
  }

  // FC head: wave w reduces batch row w of the final hidden state.
  float s = 0.f;
  const __bf16* hrow = &hbuf[cur][wave][0];
  for (int k = lane; k < H_; k += 32)
    s += (float)hrow[k] * fcw[k];
  #pragma unroll
  for (int off = 16; off > 0; off >>= 1)
    s += __shfl_xor(s, off, 32);
  if (lane == 0) out[bbase + wave] = s + fcb[0];
}

// ---------------------------------------------------------------------------
extern "C" void kernel_launch(void* const* d_in, const int* in_sizes, int n_in,
                              void* d_out, int out_size, void* d_ws, size_t ws_size,
                              hipStream_t stream) {
  (void)in_sizes; (void)n_in; (void)out_size; (void)ws_size;
  const float* x   = (const float*)d_in[0];
  const float* Wih = (const float*)d_in[1];
  const float* Whh = (const float*)d_in[2];
  const float* bih = (const float*)d_in[3];
  const float* bhh = (const float*)d_in[4];
  const float* fcw = (const float*)d_in[5];
  const float* fcb = (const float*)d_in[6];
  float* out = (float*)d_out;

  // Workspace layout (256B-aligned sections):
  char* w = (char*)d_ws;
  float*  bsum  = (float*)(w);                               //   2 KB
  __bf16* wp_ih = (__bf16*)(w + 2048);                       // 128 KB
  __bf16* wp_hh = (__bf16*)(w + 2048 + 131072);              // 512 KB
  __bf16* xp    = (__bf16*)(w + 657408);                     //  16 MB (packed x, A layout)
  float*  xs    = (float*)(w + 657408 + 16777216);           // 128 MB

  hipLaunchKernelGGL(bias_kernel,        dim3(2),     dim3(256), 0, stream, bih, bhh, bsum);
  hipLaunchKernelGGL((pack_b_kernel<4>), dim3(256),   dim3(256), 0, stream, Wih, wp_ih, 512 * 128);
  hipLaunchKernelGGL((pack_b_kernel<16>),dim3(1024),  dim3(256), 0, stream, Whh, wp_hh, 512 * 512);
  hipLaunchKernelGGL((pack_a_kernel<4>), dim3(32768), dim3(256), 0, stream, x, xp, B_ * T_ * I_);
  // 4096 M-tiles x 32 N-tiles, 8 waves/block -> 16384 blocks
  hipLaunchKernelGGL(xs_gemm_kernel, dim3(16384), dim3(256), 0, stream, xp, wp_ih, bsum, xs);
  // 16 persistent blocks (one per 16-row batch tile), 16 waves each
  hipLaunchKernelGGL(rnn_kernel, dim3(16), dim3(512), 0, stream, xs, wp_hh, fcw, fcb, out);
}
